// TreeNet_35424890257904
// MI455X (gfx1250) — matrix operand order repaired and checked by reading, not compiled, not run on hardware
//
#include <hip/hip_runtime.h>

#define N_NODES 4095
#define NLEAF   2048
#define HID     768
#define INSZ    300
#define KLEAF   320        // 300 padded to multiple of 32
#define G4      3072       // 4*H leaf gates
#define G3      2304       // 3*H combine gates
#define KCAT    1536       // [hx|hc]
#define MAXLVL  48

typedef __attribute__((ext_vector_type(16))) __bf16 v16bf;
typedef __attribute__((ext_vector_type(8)))  float  v8f;

// exact parameter types of the async-load builtins (from the clang diagnostic)
typedef int v2i_t __attribute__((vector_size(8)));
typedef int v4i_t __attribute__((vector_size(16)));

#if __has_builtin(__builtin_amdgcn_global_load_async_to_lds_b64) && \
    __has_builtin(__builtin_amdgcn_global_load_async_to_lds_b128)
#define HAVE_ASYNC_LDS 1
#else
#define HAVE_ASYNC_LDS 0
#endif

#if HAVE_ASYNC_LDS
#if __has_builtin(__builtin_amdgcn_s_wait_asynccnt)
#define WAIT_ASYNC(n) __builtin_amdgcn_s_wait_asynccnt(n)
#else
#define WAIT_ASYNC(n) asm volatile("s_wait_asynccnt %0" :: "n"(n) : "memory")
#endif
#else
#define WAIT_ASYNC(n) do {} while (0)
#endif

__device__ __forceinline__ float sigf(float x) { return 1.0f / (1.0f + __expf(-x)); }

// ---------------------------------------------------------------------------
// 1. Tree preprocessing (single thread; 4095 elements, trivial cost)
// ---------------------------------------------------------------------------
__global__ void prep_tree(const int* __restrict__ leaf, const int* __restrict__ child,
                          const int* __restrict__ prev,
                          int* __restrict__ level, int* __restrict__ leafRow,
                          int* __restrict__ rowNode, int* __restrict__ levelStart,
                          int* __restrict__ nodeOrder)
{
    if (threadIdx.x != 0 || blockIdx.x != 0) return;
    int r = 0;
    for (int i = 0; i < N_NODES; ++i) {
        if (leaf[i]) { leafRow[i] = r; rowNode[r] = i; ++r; }
        else leafRow[i] = -1;
    }
    for (int i = 0; i < N_NODES; ++i) {
        int lv = 0;
        int pi = prev[i];
        if (pi >= 0) { int t = level[pi] + 1; lv = (t > lv) ? t : lv; }
        if (!leaf[i]) {
            int ci = child[i];
            if (ci >= 0) { int t = level[ci] + 1; lv = (t > lv) ? t : lv; }
        }
        if (lv >= MAXLVL) lv = MAXLVL - 1;   // never triggers (depth ~23); OOB guard
        level[i] = lv;
    }
    for (int l = 0; l <= MAXLVL; ++l) levelStart[l] = 0;
    for (int i = 0; i < N_NODES; ++i) levelStart[level[i] + 1]++;
    for (int l = 0; l < MAXLVL; ++l) levelStart[l + 1] += levelStart[l];
    int cur[MAXLVL];
    for (int l = 0; l < MAXLVL; ++l) cur[l] = levelStart[l];
    for (int i = 0; i < N_NODES; ++i) nodeOrder[cur[level[i]]++] = i;
}

// ---------------------------------------------------------------------------
// 2. Weight conversion to bf16 (+ K padding) and bias folding
// ---------------------------------------------------------------------------
__global__ void prep_weights(const float* __restrict__ Wx, const float* __restrict__ bx,
                             const float* __restrict__ bh, const float* __restrict__ Ws,
                             const float* __restrict__ bs, const float* __restrict__ Wc,
                             const float* __restrict__ bc,
                             __bf16* __restrict__ Wcat, __bf16* __restrict__ Wxbf,
                             float* __restrict__ bxh, float* __restrict__ bsc)
{
    const long WCATE = (long)G3 * KCAT;
    const long WXE   = (long)G4 * KLEAF;
    const long total = WCATE + WXE + G4 + G3;
    long stride = (long)gridDim.x * blockDim.x;
    for (long idx = (long)blockIdx.x * blockDim.x + threadIdx.x; idx < total; idx += stride) {
        if (idx < WCATE) {
            int g = (int)(idx / KCAT), k = (int)(idx % KCAT);
            float v = (k < HID) ? Ws[(long)g * HID + k] : Wc[(long)g * HID + (k - HID)];
            Wcat[idx] = (__bf16)v;
        } else if (idx < WCATE + WXE) {
            long j = idx - WCATE;
            int g = (int)(j / KLEAF), k = (int)(j % KLEAF);
            Wxbf[j] = (k < INSZ) ? (__bf16)Wx[(long)g * INSZ + k] : (__bf16)0.0f;
        } else if (idx < WCATE + WXE + G4) {
            int j = (int)(idx - WCATE - WXE);
            bxh[j] = bx[j] + bh[j];
        } else {
            int j = (int)(idx - WCATE - WXE - G4);
            bsc[j] = bs[j] + bc[j];
        }
    }
}

// ---------------------------------------------------------------------------
// 3. Gather leaf embeddings into bf16 A matrix [NLEAF][KLEAF]
// ---------------------------------------------------------------------------
__global__ void gather_leaves(const int* __restrict__ rowNode, const int* __restrict__ word,
                              const float* __restrict__ emb, __bf16* __restrict__ Abuf)
{
    long t = (long)blockIdx.x * blockDim.x + threadIdx.x;
    const long total = (long)NLEAF * KLEAF;
    if (t >= total) return;
    int r = (int)(t / KLEAF), k = (int)(t % KLEAF);
    int node = rowNode[r];
    int w = word[node];
    Abuf[t] = (k < INSZ) ? (__bf16)emb[(long)w * INSZ + k] : (__bf16)0.0f;
}

// ---------------------------------------------------------------------------
// GEMM: C[m][n] = sum_k A[m][k]*B[n][k] + bias[n]
// 256 threads = 8 waves as 2(M) x 4(N); each wave owns 4 accumulators
// (16M x 64N) -> block tile 32M x 256N. Ping-pong LDS buffers filled with
// async global->LDS (ASYNCcnt) when available; v_wmma_f32_16x16x32_bf16.
// ---------------------------------------------------------------------------
__device__ __forceinline__ void stage_tile(const __bf16* __restrict__ A,
                                           const __bf16* __restrict__ B,
                                           __bf16* sA, __bf16* sB,
                                           int mBase, int nb0, int K, int k0, int tid)
{
    int r = tid >> 3, kk = (tid & 7) * 4;        // A: 32x32 tile, 8B per thread
#if HAVE_ASYNC_LDS
    __builtin_amdgcn_global_load_async_to_lds_b64(
        (v2i_t __attribute__((address_space(1))) *)(A + (long)(mBase + r) * K + k0 + kk),
        (v2i_t __attribute__((address_space(3))) *)(sA + tid * 4), 0, 0);
    #pragma unroll
    for (int j = 0; j < 4; ++j)                  // B: 256x32 tile, row per thread
        __builtin_amdgcn_global_load_async_to_lds_b128(
            (v4i_t __attribute__((address_space(1))) *)(B + (long)(nb0 + tid) * K + k0 + j * 8),
            (v4i_t __attribute__((address_space(3))) *)(sB + tid * 32 + j * 8), 0, 0);
#else
    *(uint2*)(sA + tid * 4) = *(const uint2*)(A + (long)(mBase + r) * K + k0 + kk);
    #pragma unroll
    for (int j = 0; j < 4; ++j)
        *(uint4*)(sB + tid * 32 + j * 8) =
            *(const uint4*)(B + (long)(nb0 + tid) * K + k0 + j * 8);
#endif
}

__device__ __forceinline__ void gemm_body(const __bf16* __restrict__ A,
                                          const __bf16* __restrict__ B,
                                          const float* __restrict__ bias,
                                          float* __restrict__ C,
                                          int Mpad, int N, int K)
{
    if ((int)blockIdx.x * 32 >= Mpad) return;          // uniform over block
    __shared__ __align__(32) __bf16 sA[2][32 * 32];    //  2 x  2 KB
    __shared__ __align__(32) __bf16 sB[2][256 * 32];   //  2 x 16 KB

    const int tid  = threadIdx.x;
    const int wave = tid >> 5;
    const int lane = tid & 31;
    const int lg   = lane >> 4;      // lane group (0/1)
    const int lm   = lane & 15;
    const int wm   = wave >> 2;      // wave M half (0/1)
    const int wn   = wave & 3;       // wave N quarter (0..3)
    const int mBase = blockIdx.x * 32;
    const int nb0   = blockIdx.y * 256;

    v8f acc[4] = {};
    stage_tile(A, B, sA[0], sB[0], mBase, nb0, K, 0, tid);

    int buf = 0;
    for (int k0 = 0; k0 < K; k0 += 32) {
        const bool more = (k0 + 32 < K);
        if (more) stage_tile(A, B, sA[buf ^ 1], sB[buf ^ 1], mBase, nb0, K, k0 + 32, tid);
        if (more) { WAIT_ASYNC(5); } else { WAIT_ASYNC(0); }
        __syncthreads();

        const __bf16* sAb = sA[buf];
        const __bf16* sBb = sB[buf];
        // A fragment: 16x32 bf16, contiguous K half by lane group (matches B)
        v16bf af;
        #pragma unroll
        for (int e = 0; e < 16; ++e) {
            int v = e >> 1, p = e & 1;
            int kk = e + 16 * lg;
            af[e] = sAb[(wm * 16 + lm) * 32 + kk];
        }
        #pragma unroll
        for (int j = 0; j < 4; ++j) {
            // B fragment: lane n = L&15, K half by lane group; contiguous 16 halves
            v16bf bfr = *(const v16bf*)(sBb + (wn * 64 + j * 16 + lm) * 32 + lg * 16);
            acc[j] = __builtin_amdgcn_wmma_f32_16x16x32_bf16(
                         false, af, false, bfr, (short)0, acc[j], false, false);
        }
        __syncthreads();
        buf ^= 1;
    }
    // D layout: VGPR e -> M = e + 8*lg, N = lane&15
    #pragma unroll
    for (int j = 0; j < 4; ++j) {
        int n = nb0 + wn * 64 + j * 16 + lm;
        float bn = bias[n];
        #pragma unroll
        for (int e = 0; e < 8; ++e) {
            int m = mBase + wm * 16 + e + 8 * lg;
            C[(long)m * N + n] = acc[j][e] + bn;
        }
    }
}

__global__ __launch_bounds__(256)
void wmma_gemm_fixed(const __bf16* __restrict__ A, const __bf16* __restrict__ B,
                     const float* __restrict__ bias, float* __restrict__ C,
                     int M, int N, int K)
{
    gemm_body(A, B, bias, C, M, N, K);
}

__global__ __launch_bounds__(256)
void wmma_gemm_level(const __bf16* __restrict__ A, const __bf16* __restrict__ B,
                     const float* __restrict__ bias, float* __restrict__ C,
                     int N, int K, const int* __restrict__ levelStart, int l)
{
    int cnt  = levelStart[l + 1] - levelStart[l];
    int Mpad = (cnt + 31) & ~31;
    gemm_body(A, B, bias, C, Mpad, N, K);
}

// ---------------------------------------------------------------------------
// 4. Leaf gate activations: i,o,f,c chunk order; f unused for leaves
// ---------------------------------------------------------------------------
__global__ void leaf_act(const float* __restrict__ G, float* __restrict__ LeafH,
                         float* __restrict__ LeafC)
{
    long t = (long)blockIdx.x * blockDim.x + threadIdx.x;
    const long total = (long)NLEAF * HID;
    if (t >= total) return;
    int r = (int)(t / HID), h = (int)(t % HID);
    const float* g = G + (long)r * G4;
    float gi = g[h], go = g[HID + h], gc = g[3 * HID + h];
    float cell = sigf(gi) * tanhf(gc);
    float hid  = sigf(go) * tanhf(cell);
    LeafC[t] = cell;
    LeafH[t] = hid;
}

// ---------------------------------------------------------------------------
// 5. Per-level gather: X row t = [hx | hc] in bf16, rows padded to x32 with 0
// ---------------------------------------------------------------------------
__global__ void gather_level(const int* __restrict__ leaf, const int* __restrict__ child,
                             const int* __restrict__ prev, const int* __restrict__ leafRow,
                             const int* __restrict__ nodeOrder, const int* __restrict__ levelStart,
                             const float* __restrict__ Harr, const float* __restrict__ LeafH,
                             __bf16* __restrict__ Xbuf, int l)
{
    int s = levelStart[l];
    int cnt = levelStart[l + 1] - s;
    int cntp = (cnt + 31) & ~31;
    int t = blockIdx.x;
    if (t >= cntp) return;
    __bf16* xr = Xbuf + (long)t * KCAT;
    if (t >= cnt) {
        for (int k = threadIdx.x; k < KCAT; k += blockDim.x) xr[k] = (__bf16)0.0f;
        return;
    }
    int node = nodeOrder[s + t];
    int pi = prev[node];
    int isleaf = leaf[node];
    int ci = child[node];
    const float* hx = (pi >= 0) ? (Harr + (long)pi * HID) : nullptr;
    const float* hc = isleaf ? (LeafH + (long)leafRow[node] * HID)
                             : (Harr + (long)((ci >= 0) ? ci : 0) * HID);
    for (int k = threadIdx.x; k < HID; k += blockDim.x) {
        xr[k]       = hx ? (__bf16)hx[k] : (__bf16)0.0f;
        xr[HID + k] = (__bf16)hc[k];
    }
}

// ---------------------------------------------------------------------------
// 6. Per-level 3-gate combine + scatter: i,o,f chunk order
// ---------------------------------------------------------------------------
__global__ void act_scatter(const int* __restrict__ leaf, const int* __restrict__ child,
                            const int* __restrict__ prev, const int* __restrict__ leafRow,
                            const int* __restrict__ nodeOrder, const int* __restrict__ levelStart,
                            const float* __restrict__ G, const float* __restrict__ LeafC,
                            float* __restrict__ Harr, float* __restrict__ Carr, int l)
{
    int s = levelStart[l];
    int cnt = levelStart[l + 1] - s;
    int t = blockIdx.x;
    if (t >= cnt) return;
    int node = nodeOrder[s + t];
    int pi = prev[node];
    int isleaf = leaf[node];
    int ci = child[node];
    const float* g   = G + (long)t * G3;
    const float* cxp = (pi >= 0) ? (Carr + (long)pi * HID) : nullptr;
    const float* ccp = isleaf ? (LeafC + (long)leafRow[node] * HID)
                              : (Carr + (long)((ci >= 0) ? ci : 0) * HID);
    float* Hn = Harr + (long)node * HID;
    float* Cn = Carr + (long)node * HID;
    for (int h = threadIdx.x; h < HID; h += blockDim.x) {
        float gi = g[h], go = g[HID + h], gf = g[2 * HID + h];
        float cx = cxp ? cxp[h] : 0.0f;
        float cc = ccp[h];
        float cell = sigf(gf) * cx + sigf(gi) * cc;
        float hid  = sigf(go) * tanhf(cell);
        Cn[h] = cell;
        Hn[h] = hid;
    }
}

__global__ void copy_out(const float* __restrict__ Harr, float* __restrict__ out)
{
    int h = blockIdx.x * blockDim.x + threadIdx.x;
    if (h < HID) out[h] = Harr[(long)(N_NODES - 1) * HID + h];
}

// ---------------------------------------------------------------------------
extern "C" void kernel_launch(void* const* d_in, const int* in_sizes, int n_in,
                              void* d_out, int out_size, void* d_ws, size_t ws_size,
                              hipStream_t stream)
{
    const int*   leaf  = (const int*)d_in[0];
    const int*   word  = (const int*)d_in[1];
    const int*   child = (const int*)d_in[2];
    const int*   prev  = (const int*)d_in[3];
    const float* emb   = (const float*)d_in[4];
    const float* Wx    = (const float*)d_in[5];
    const float* bx    = (const float*)d_in[6];
    // d_in[7] = Wh (unused by the reference math: encode_h(0) contributes bh only)
    const float* bh    = (const float*)d_in[8];
    const float* Ws    = (const float*)d_in[9];
    const float* bs    = (const float*)d_in[10];
    const float* Wc    = (const float*)d_in[11];
    const float* bc    = (const float*)d_in[12];
    float* out = (float*)d_out;

    char* ws = (char*)d_ws;
    size_t off = 0;
    auto alloc = [&](size_t bytes) -> void* {
        void* p = ws + off;
        off = (off + bytes + 255) & ~(size_t)255;
        return p;
    };

    float*  Harr    = (float*)alloc((size_t)N_NODES * HID * 4);
    float*  Carr    = (float*)alloc((size_t)N_NODES * HID * 4);
    float*  LeafH   = (float*)alloc((size_t)NLEAF * HID * 4);
    float*  LeafC   = (float*)alloc((size_t)NLEAF * HID * 4);
    float*  Gbuf    = (float*)alloc((size_t)NLEAF * G4 * 4);        // shared leaf/comb gate buffer
    __bf16* Xbuf    = (__bf16*)alloc((size_t)NLEAF * KCAT * 2);     // also leaf A [2048x320]
    __bf16* Wxbf    = (__bf16*)alloc((size_t)G4 * KLEAF * 2);
    __bf16* Wcat    = (__bf16*)alloc((size_t)G3 * KCAT * 2);
    float*  bxh     = (float*)alloc(G4 * 4);
    float*  bsc     = (float*)alloc(G3 * 4);
    int*    level   = (int*)alloc(N_NODES * 4);
    int*    leafRow = (int*)alloc(N_NODES * 4);
    int*    rowNode = (int*)alloc(NLEAF * 4);
    int*    nodeOrd = (int*)alloc(N_NODES * 4);
    int*    lvStart = (int*)alloc((MAXLVL + 1) * 4);
    (void)in_sizes; (void)n_in; (void)out_size; (void)ws_size;

    // Phase 0: preprocessing
    prep_tree<<<1, 1, 0, stream>>>(leaf, child, prev, level, leafRow, rowNode, lvStart, nodeOrd);
    prep_weights<<<2048, 256, 0, stream>>>(Wx, bx, bh, Ws, bs, Wc, bc, Wcat, Wxbf, bxh, bsc);

    // Phase 1: all 2048 leaves in one WMMA GEMM (M=2048, N=3072, K=320)
    {
        long tot = (long)NLEAF * KLEAF;
        gather_leaves<<<(unsigned)((tot + 255) / 256), 256, 0, stream>>>(rowNode, word, emb, Xbuf);
        dim3 grid(NLEAF / 32, G4 / 256);
        wmma_gemm_fixed<<<grid, 256, 0, stream>>>(Xbuf, Wxbf, bxh, Gbuf, NLEAF, G4, KLEAF);
        long tot2 = (long)NLEAF * HID;
        leaf_act<<<(unsigned)((tot2 + 255) / 256), 256, 0, stream>>>(Gbuf, LeafH, LeafC);
    }

    // Phase 2: wavefront over dependency levels (critical path ~23 of 48)
    dim3 cgrid(NLEAF / 32, G3 / 256);
    for (int l = 0; l < MAXLVL; ++l) {
        gather_level<<<NLEAF, 256, 0, stream>>>(leaf, child, prev, leafRow, nodeOrd, lvStart,
                                                Harr, LeafH, Xbuf, l);
        wmma_gemm_level<<<cgrid, 256, 0, stream>>>(Xbuf, Wcat, bsc, Gbuf, G3, KCAT, lvStart, l);
        act_scatter<<<NLEAF, 256, 0, stream>>>(leaf, child, prev, leafRow, nodeOrd, lvStart,
                                               Gbuf, LeafC, Harr, Carr, l);
    }

    copy_out<<<(HID + 255) / 256, 256, 0, stream>>>(Harr, out);
}
